// GNNLayer_75986561401309
// MI455X (gfx1250) — matrix operand order repaired and checked by reading, not compiled, not run on hardware
//
#include <hip/hip_runtime.h>
#include <cstdint>

typedef __attribute__((ext_vector_type(16))) __bf16 v16bf;
typedef __attribute__((ext_vector_type(8)))  float  v8f;

#define N_NODES 100000
#define N_EDGES 800000

#define WPB  2     // waves per block
#define LDA1 200   // 192 + 8 pad (floats)
#define LDX  136   // 128 + 8 pad (floats)

static __device__ inline v8f wmma_bf16(v16bf a, v16bf b, v8f c) {
  // 8 args: (neg_a, A, neg_b, B, c_mod, C, reuse_a, reuse_b)
  return __builtin_amdgcn_wmma_f32_16x16x32_bf16(false, a, false, b, (short)0, c,
                                                 false, false);
}

// Build A (and its bf16 residual) fragment for one 16x32 K-slab from an LDS row.
// Contiguous k order used here (16-bit A 16x32): lane row = lane%16; element e -> K = e + 16*hiL.
static __device__ inline void build_afrag(const float* rowp, int hiL,
                                          v16bf& ahi, v16bf& alo) {
  const int b0 = 16 * hiL;
  const int b1 = 8 + 16 * hiL;
#pragma unroll
  for (int e = 0; e < 8; ++e) {
    float x0 = rowp[b0 + e];
    __bf16 h0 = (__bf16)x0;
    ahi[e]     = h0;
    alo[e]     = (__bf16)(x0 - (float)h0);
    float x1 = rowp[b1 + e];
    __bf16 h1 = (__bf16)x1;
    ahi[e + 8] = h1;
    alo[e + 8] = (__bf16)(x1 - (float)h1);
  }
}

// 16-row GEMM: C(16 x NTILES*16) += A(16 x KSTEPS*32) * W(K x N), W given
// transposed [N][K] in bf16 hi/lo. B frag: lane col = lane%16, element e ->
// K = e + 16*hiL (contiguous 32B load from transposed weights).
template <int KSTEPS, int NTILES>
static __device__ inline void mm16(const float* Abuf, int lda,
                                   const __bf16* __restrict__ Whi,
                                   const __bf16* __restrict__ Wlo,
                                   v8f* acc, int loL, int hiL) {
  constexpr int K = KSTEPS * 32;
#pragma unroll
  for (int ks = 0; ks < KSTEPS; ++ks) {
    const int kb = ks * 32;
    v16bf ahi, alo;
    build_afrag(Abuf + loL * lda + kb, hiL, ahi, alo);
#pragma unroll
    for (int nt = 0; nt < NTILES; ++nt) {
      const int boff = (nt * 16 + loL) * K + kb + 16 * hiL;
      v16bf bhi = *(const v16bf*)(Whi + boff);
      v16bf blo = *(const v16bf*)(Wlo + boff);
      acc[nt] = wmma_bf16(ahi, bhi, acc[nt]);  // main product
      acc[nt] = wmma_bf16(alo, bhi, acc[nt]);  // A residual
      acc[nt] = wmma_bf16(ahi, blo, acc[nt]);  // B residual
    }
  }
}

// Transpose f32 [K][N] weights into bf16 hi + bf16 residual, stored [N][K].
__global__ void gnn_prep_kernel(const float* __restrict__ W,
                                __bf16* __restrict__ hi,
                                __bf16* __restrict__ lo, int K, int N) {
  int idx = blockIdx.x * blockDim.x + threadIdx.x;
  if (idx >= K * N) return;
  int k = idx / N, n = idx % N;
  float x = W[idx];
  __bf16 hv = (__bf16)x;
  hi[n * K + k] = hv;
  lo[n * K + k] = (__bf16)(x - (float)hv);
}

// Fused edge pipeline: gather -> msgMLP -> mask -> atomic scatter -> edgeMLP.
__global__ __launch_bounds__(WPB * 32) void gnn_edge_kernel(
    const float* __restrict__ h, const float* __restrict__ e,
    const int* __restrict__ snd, const int* __restrict__ rcv,
    const float* __restrict__ mask,
    const __bf16* __restrict__ Wm1h, const __bf16* __restrict__ Wm1l,
    const float* __restrict__ bm1,
    const __bf16* __restrict__ Wm2h, const __bf16* __restrict__ Wm2l,
    const float* __restrict__ bm2,
    const __bf16* __restrict__ We1h, const __bf16* __restrict__ We1l,
    const float* __restrict__ be1,
    const __bf16* __restrict__ We2h, const __bf16* __restrict__ We2l,
    const float* __restrict__ be2,
    float* __restrict__ agg, float* __restrict__ e_new) {
  __shared__ float Abuf[WPB][16][LDA1];
  __shared__ float Xbuf[WPB][16][LDX];
  __shared__ int   snd_s[WPB][16];
  __shared__ int   rcv_s[WPB][16];
  __shared__ float msk_s[WPB][16];

  const int w    = threadIdx.x >> 5;
  const int lane = threadIdx.x & 31;
  const int loL  = lane & 15;
  const int hiL  = lane >> 4;
  const size_t eb = ((size_t)blockIdx.x * WPB + w) * 16;

  if (lane < 16) {
    snd_s[w][lane] = snd[eb + lane];
    rcv_s[w][lane] = rcv[eb + lane];
    msk_s[w][lane] = mask[eb + lane];
  }
  __syncthreads();

  // Gather [ e | h[snd] | h[rcv] ] -> Abuf (16 x 192), float4 per step.
#pragma unroll 4
  for (int idx = lane; idx < 16 * 48; idx += 32) {
    int r  = idx / 48;
    int c4 = (idx % 48) * 4;
    const float* src;
    if (c4 < 64)       src = e + (eb + r) * 64 + c4;
    else if (c4 < 128) src = h + (size_t)snd_s[w][r] * 64 + (c4 - 64);
    else               src = h + (size_t)rcv_s[w][r] * 64 + (c4 - 128);
    *(float4*)&Abuf[w][r][c4] = *(const float4*)src;
  }
  __syncthreads();

  // GEMM1: (16x192)x(192x128) + bias, relu -> Xbuf
  v8f acc8[8] = {};
  mm16<6, 8>(&Abuf[w][0][0], LDA1, Wm1h, Wm1l, acc8, loL, hiL);
#pragma unroll
  for (int nt = 0; nt < 8; ++nt) {
    float bv = bm1[nt * 16 + loL];
#pragma unroll
    for (int i = 0; i < 8; ++i) {
      float x = acc8[nt][i] + bv;
      Xbuf[w][i + 8 * hiL][nt * 16 + loL] = x > 0.f ? x : 0.f;
    }
  }
  __syncthreads();

  // GEMM2: (16x128)x(128x64) + bias -> m; mask; atomic scatter; m -> Abuf[:,64:128]
  v8f acc4[4] = {};
  mm16<4, 4>(&Xbuf[w][0][0], LDX, Wm2h, Wm2l, acc4, loL, hiL);
#pragma unroll
  for (int nt = 0; nt < 4; ++nt) {
    float bv = bm2[nt * 16 + loL];
#pragma unroll
    for (int i = 0; i < 8; ++i) {
      int r = i + 8 * hiL;
      float mval = (acc4[nt][i] + bv) * msk_s[w][r];
      atomicAdd(&agg[(size_t)rcv_s[w][r] * 64 + nt * 16 + loL], mval);
      Abuf[w][r][64 + nt * 16 + loL] = mval;
    }
  }
  __syncthreads();

  // GEMM3: [e|m](16x128)x(128x128) + bias, relu -> Xbuf
#pragma unroll
  for (int nt = 0; nt < 8; ++nt) acc8[nt] = (v8f){};
  mm16<4, 8>(&Abuf[w][0][0], LDA1, We1h, We1l, acc8, loL, hiL);
#pragma unroll
  for (int nt = 0; nt < 8; ++nt) {
    float bv = be1[nt * 16 + loL];
#pragma unroll
    for (int i = 0; i < 8; ++i) {
      float x = acc8[nt][i] + bv;
      Xbuf[w][i + 8 * hiL][nt * 16 + loL] = x > 0.f ? x : 0.f;
    }
  }
  __syncthreads();

  // GEMM4: (16x128)x(128x64) + bias -> e_new
#pragma unroll
  for (int nt = 0; nt < 4; ++nt) acc4[nt] = (v8f){};
  mm16<4, 4>(&Xbuf[w][0][0], LDX, We2h, We2l, acc4, loL, hiL);
#pragma unroll
  for (int nt = 0; nt < 4; ++nt) {
    float bv = be2[nt * 16 + loL];
#pragma unroll
    for (int i = 0; i < 8; ++i)
      e_new[(eb + i + 8 * hiL) * 64 + nt * 16 + loL] = acc4[nt][i] + bv;
  }
}

// Node update: [h | agg] -> MLP -> h_new
__global__ __launch_bounds__(WPB * 32) void gnn_node_kernel(
    const float* __restrict__ h, const float* __restrict__ agg,
    const __bf16* __restrict__ Wn1h, const __bf16* __restrict__ Wn1l,
    const float* __restrict__ bn1,
    const __bf16* __restrict__ Wn2h, const __bf16* __restrict__ Wn2l,
    const float* __restrict__ bn2, float* __restrict__ h_new) {
  __shared__ float Abuf[WPB][16][LDX];
  __shared__ float Xbuf[WPB][16][LDX];

  const int w    = threadIdx.x >> 5;
  const int lane = threadIdx.x & 31;
  const int loL  = lane & 15;
  const int hiL  = lane >> 4;
  const size_t nb = ((size_t)blockIdx.x * WPB + w) * 16;

#pragma unroll 4
  for (int idx = lane; idx < 16 * 32; idx += 32) {
    int r  = idx / 32;
    int c4 = (idx % 32) * 4;
    const float* src = (c4 < 64) ? h + (nb + r) * 64 + c4
                                 : agg + (nb + r) * 64 + (c4 - 64);
    *(float4*)&Abuf[w][r][c4] = *(const float4*)src;
  }
  __syncthreads();

  v8f acc8[8] = {};
  mm16<4, 8>(&Abuf[w][0][0], LDX, Wn1h, Wn1l, acc8, loL, hiL);
#pragma unroll
  for (int nt = 0; nt < 8; ++nt) {
    float bv = bn1[nt * 16 + loL];
#pragma unroll
    for (int i = 0; i < 8; ++i) {
      float x = acc8[nt][i] + bv;
      Xbuf[w][i + 8 * hiL][nt * 16 + loL] = x > 0.f ? x : 0.f;
    }
  }
  __syncthreads();

  v8f acc4[4] = {};
  mm16<4, 4>(&Xbuf[w][0][0], LDX, Wn2h, Wn2l, acc4, loL, hiL);
#pragma unroll
  for (int nt = 0; nt < 4; ++nt) {
    float bv = bn2[nt * 16 + loL];
#pragma unroll
    for (int i = 0; i < 8; ++i)
      h_new[(nb + i + 8 * hiL) * 64 + nt * 16 + loL] = acc4[nt][i] + bv;
  }
}

extern "C" void kernel_launch(void* const* d_in, const int* in_sizes, int n_in,
                              void* d_out, int out_size, void* d_ws,
                              size_t ws_size, hipStream_t stream) {
  const float* h      = (const float*)d_in[0];
  const float* e      = (const float*)d_in[1];
  const int*   snd    = (const int*)d_in[2];
  const int*   rcv    = (const int*)d_in[3];
  const float* mask   = (const float*)d_in[4];
  const float* W_msg1 = (const float*)d_in[5];
  const float* b_msg1 = (const float*)d_in[6];
  const float* W_msg2 = (const float*)d_in[7];
  const float* b_msg2 = (const float*)d_in[8];
  const float* W_nu1  = (const float*)d_in[9];
  const float* b_nu1  = (const float*)d_in[10];
  const float* W_nu2  = (const float*)d_in[11];
  const float* b_nu2  = (const float*)d_in[12];
  const float* W_eu1  = (const float*)d_in[13];
  const float* b_eu1  = (const float*)d_in[14];
  const float* W_eu2  = (const float*)d_in[15];
  const float* b_eu2  = (const float*)d_in[16];

  char* p = (char*)d_ws;
  float* agg = (float*)p;
  p += (size_t)N_NODES * 64 * sizeof(float);
  auto take = [&](size_t elems) -> __bf16* {
    __bf16* q = (__bf16*)p;
    p += elems * sizeof(__bf16);
    return q;
  };
  __bf16 *Wm1h = take(192 * 128), *Wm1l = take(192 * 128);
  __bf16 *Wm2h = take(128 * 64),  *Wm2l = take(128 * 64);
  __bf16 *Wn1h = take(128 * 128), *Wn1l = take(128 * 128);
  __bf16 *Wn2h = take(128 * 64),  *Wn2l = take(128 * 64);
  __bf16 *We1h = take(128 * 128), *We1l = take(128 * 128);
  __bf16 *We2h = take(128 * 64),  *We2l = take(128 * 64);

  hipMemsetAsync(agg, 0, (size_t)N_NODES * 64 * sizeof(float), stream);

  auto prep = [&](const float* W, __bf16* hi, __bf16* lo, int K, int N) {
    int total = K * N;
    gnn_prep_kernel<<<(total + 255) / 256, 256, 0, stream>>>(W, hi, lo, K, N);
  };
  prep(W_msg1, Wm1h, Wm1l, 192, 128);
  prep(W_msg2, Wm2h, Wm2l, 128, 64);
  prep(W_nu1,  Wn1h, Wn1l, 128, 128);
  prep(W_nu2,  Wn2h, Wn2l, 128, 64);
  prep(W_eu1,  We1h, We1l, 128, 128);
  prep(W_eu2,  We2h, We2l, 128, 64);

  float* out_h = (float*)d_out;
  float* out_e = out_h + (size_t)N_NODES * 64;

  gnn_edge_kernel<<<N_EDGES / (16 * WPB), WPB * 32, 0, stream>>>(
      h, e, snd, rcv, mask, Wm1h, Wm1l, b_msg1, Wm2h, Wm2l, b_msg2, We1h, We1l,
      b_eu1, We2h, We2l, b_eu2, agg, out_e);

  gnn_node_kernel<<<N_NODES / (16 * WPB), WPB * 32, 0, stream>>>(
      h, agg, Wn1h, Wn1l, b_nu1, Wn2h, Wn2l, b_nu2, out_h);
}
